// ProtoNet_2869038153783
// MI455X (gfx1250) — matrix operand order repaired and checked by reading, not compiled, not run on hardware
//
#include <hip/hip_runtime.h>
#include <hip/hip_bf16.h>

// ProtoNet (PointNet++ backbone) for MI455X / gfx1250.
// All 12 GEMMs (9 SA-MLP layers + 3 FC layers, ~135 GFLOP across the two
// embedding passes) run through v_wmma_f32_16x16x32_f16 with fp32 accum.
// GEMM is register-blocked: one wave = 64x32 output patch (8 WMMAs per
// K-step from 4 A-frags + 2 B-frags => ~21 FLOP/byte of load traffic).
// Irregular stages (FPS, ball query, BN stats, max-pool, proto logits)
// are plain wave32 kernels.
//
// Workspace (carved from d_ws, ~460 MB peak, fully reused):
//   Abuf: half activations (max 327680 x 160) | Cbuf: fp32 pre-acts
//   (max 655360 x 128) | Wh: half packed weights | stage buffers.

typedef __attribute__((ext_vector_type(16))) _Float16 v16h;
typedef __attribute__((ext_vector_type(8)))  float    v8f;

#define EPSV 1e-5f

// ---------------------------------------------------------------- WMMA GEMM
// C[r, o] = sum_k A[r,k] * W[o,k] + bias[o]
// A: rowsPad x K (half, K mult of 32), W: cout x K (half), C: rows x cout.
// Block = 128 threads = 4 waves. Wave computes 4 M-tiles x 2 N-tiles.
// Block covers 256 rows x 32 cols. cout must be a multiple of 32.
__global__ __launch_bounds__(128) void k_gemm64x32(
    const _Float16* __restrict__ A, const _Float16* __restrict__ W,
    const float* __restrict__ bias, float* __restrict__ C,
    int rows, int rowsPad, int K, int cout) {
  const int wave = threadIdx.x >> 5;
  const int lane = threadIdx.x & 31;
  const int mtbase = blockIdx.x * 16 + wave * 4;     // first of 4 M tiles
  if (mtbase * 16 >= rowsPad) return;

  const int kband = (lane >> 4) * 16;                // A frag: 16 contig K
  const int kcont = (lane >> 4) * 16;                // B frag: 16 contig K
  const int ncol0 = blockIdx.y * 32 + (lane & 15);   // first N sub-tile col
  const int ncol1 = ncol0 + 16;

  // Per-sub-tile validity is wave-uniform => EXEC stays all-ones for WMMA.
  bool mv[4];
  const _Float16* __restrict__ Arow[4];
#pragma unroll
  for (int j = 0; j < 4; ++j) {
    const int mt = mtbase + j;
    mv[j] = (mt * 16 < rowsPad);
    const int mrow = mt * 16 + (lane & 15);
    Arow[j] = A + (size_t)(mv[j] ? mrow : 0) * K;
  }
  const _Float16* __restrict__ Wrow0 = W + (size_t)ncol0 * K;
  const _Float16* __restrict__ Wrow1 = W + (size_t)ncol1 * K;

  v8f acc[4][2];
#pragma unroll
  for (int j = 0; j < 4; ++j) {
    acc[j][0] = (v8f){};
    acc[j][1] = (v8f){};
  }

  for (int k0 = 0; k0 < K; k0 += 32) {
    if (k0 + 64 < K) {
      __builtin_prefetch(Arow[0] + k0 + 64, 0, 1);   // global_prefetch_b8
    }
    v16h b0, b1;
#pragma unroll
    for (int i = 0; i < 16; ++i) {
      b0[i] = Wrow0[k0 + kcont + i];
      b1[i] = Wrow1[k0 + kcont + i];
    }
#pragma unroll
    for (int j = 0; j < 4; ++j) {
      if (mv[j]) {
        v16h a;
#pragma unroll
        for (int i = 0; i < 16; ++i) {
          a[i] = Arow[j][k0 + kband + i];
        }
        acc[j][0] = __builtin_amdgcn_wmma_f32_16x16x32_f16(
            false, a, false, b0, (short)0, acc[j][0], false, false);
        acc[j][1] = __builtin_amdgcn_wmma_f32_16x16x32_f16(
            false, a, false, b1, (short)0, acc[j][1], false, false);
      }
    }
  }

  const float bv0 = bias[ncol0];
  const float bv1 = bias[ncol1];
#pragma unroll
  for (int j = 0; j < 4; ++j) {
    if (!mv[j]) continue;
    const int mbase = (mtbase + j) * 16 + (lane >> 4) * 8;  // D: VGPR e -> row
#pragma unroll
    for (int e = 0; e < 8; ++e) {
      const int r = mbase + e;
      if (r < rows) {
        C[(size_t)r * cout + ncol0] = acc[j][0][e] + bv0;
        C[(size_t)r * cout + ncol1] = acc[j][1][e] + bv1;
      }
    }
  }
}

// ---------------------------------------------------------------- FPS
// One block per batch element. Matches jax.lax.scan semantics: output j is
// the carry *before* update; argmax ties -> lowest index.
__global__ __launch_bounds__(256) void k_fps(
    const float* __restrict__ xyz, int xs, int N, int npoint,
    float* __restrict__ new_xyz) {
  __shared__ float dist[1024];
  __shared__ float rv[256];
  __shared__ int   ri[256];
  __shared__ int   s_far;
  __shared__ float cx, cy, cz;

  const int b = blockIdx.x;
  const int tid = threadIdx.x;
  const float* __restrict__ base = xyz + (size_t)b * N * xs;

  for (int n = tid; n < N; n += 256) dist[n] = 1e10f;
  if (tid == 0) s_far = 0;
  __syncthreads();

  for (int j = 0; j < npoint; ++j) {
    const int far = s_far;
    if (tid == 0) {
      const float* c = base + (size_t)far * xs;
      cx = c[0]; cy = c[1]; cz = c[2];
      float* o = new_xyz + ((size_t)b * npoint + j) * 3;
      o[0] = cx; o[1] = cy; o[2] = cz;
    }
    __syncthreads();
    float bv = -1.f; int bi = 0;
    for (int n = tid; n < N; n += 256) {
      const float* p = base + (size_t)n * xs;
      const float dx = p[0] - cx, dy = p[1] - cy, dz = p[2] - cz;
      const float dm = fminf(dist[n], dx * dx + dy * dy + dz * dz);
      dist[n] = dm;
      if (dm > bv) { bv = dm; bi = n; }
    }
    rv[tid] = bv; ri[tid] = bi;
    __syncthreads();
    for (int st = 128; st > 0; st >>= 1) {
      if (tid < st) {
        if (rv[tid + st] > rv[tid] ||
            (rv[tid + st] == rv[tid] && ri[tid + st] < ri[tid])) {
          rv[tid] = rv[tid + st]; ri[tid] = ri[tid + st];
        }
      }
      __syncthreads();
    }
    if (tid == 0) s_far = ri[0];
    __syncthreads();
  }
}

// ---------------------------------------------------------------- ball query + group
__device__ inline void write_group_row(_Float16* __restrict__ rowp,
                                       const float* __restrict__ xp,
                                       float cx, float cy, float cz,
                                       const float* __restrict__ pp,
                                       int Cp, int Kpad) {
  rowp[0] = (_Float16)(xp[0] - cx);
  rowp[1] = (_Float16)(xp[1] - cy);
  rowp[2] = (_Float16)(xp[2] - cz);
  for (int c = 0; c < Cp; ++c) rowp[3 + c] = (_Float16)pp[c];
  for (int c = 3 + Cp; c < Kpad; ++c) rowp[c] = (_Float16)0.f;
}

__global__ void k_ball_group(const float* __restrict__ xyz, int xs,
                             const float* __restrict__ pts, int ps, int Cp,
                             const float* __restrict__ new_xyz,
                             _Float16* __restrict__ A, float r2,
                             int total, int N, int S, int nsample, int Kpad) {
  const int t = blockIdx.x * blockDim.x + threadIdx.x;
  if (t >= total) return;
  const int b = t / S;
  const float* nx = new_xyz + (size_t)t * 3;
  const float cx = nx[0], cy = nx[1], cz = nx[2];
  const float* __restrict__ xb = xyz + (size_t)b * N * xs;
  const float* __restrict__ pb = pts + (size_t)b * N * ps;
  _Float16* __restrict__ rowBase = A + (size_t)t * nsample * Kpad;

  int cnt = 0, first = -1;
  for (int n = 0; n < N && cnt < nsample; ++n) {
    const float* xp = xb + (size_t)n * xs;
    const float dx = xp[0] - cx, dy = xp[1] - cy, dz = xp[2] - cz;
    if (dx * dx + dy * dy + dz * dz <= r2) {
      if (first < 0) first = n;
      write_group_row(rowBase + (size_t)cnt * Kpad, xp, cx, cy, cz,
                      pb + (size_t)n * ps, Cp, Kpad);
      ++cnt;
    }
  }
  if (first < 0) first = N - 1;  // degenerate: mimic clamped OOB gather
  for (; cnt < nsample; ++cnt)
    write_group_row(rowBase + (size_t)cnt * Kpad, xb + (size_t)first * xs,
                    cx, cy, cz, pb + (size_t)first * ps, Cp, Kpad);
}

// ---------------------------------------------------------------- group-all concat
__global__ void k_groupall(const float* __restrict__ l2x,
                           const float* __restrict__ l2p,
                           _Float16* __restrict__ A,
                           int rows, int Cp, int Kpad) {
  const int t = blockIdx.x * blockDim.x + threadIdx.x;
  if (t >= rows) return;
  _Float16* rowp = A + (size_t)t * Kpad;
  const float* x = l2x + (size_t)t * 3;
  const float* p = l2p + (size_t)t * Cp;
  rowp[0] = (_Float16)x[0]; rowp[1] = (_Float16)x[1]; rowp[2] = (_Float16)x[2];
  for (int c = 0; c < Cp; ++c) rowp[3 + c] = (_Float16)p[c];
  for (int c = 3 + Cp; c < Kpad; ++c) rowp[c] = (_Float16)0.f;
}

// ---------------------------------------------------------------- helpers
__global__ void k_zero(float* __restrict__ p, int n) {
  for (int i = blockIdx.x * blockDim.x + threadIdx.x; i < n;
       i += gridDim.x * blockDim.x) p[i] = 0.f;
}

// fp32 (rows x cols) -> half (rowsPad x Kpad), zero-padded. Also packs weights.
__global__ void k_pack_half(const float* __restrict__ S, _Float16* __restrict__ A,
                            int rows, int rowsPad, int cols, int Kpad) {
  const size_t total = (size_t)rowsPad * Kpad;
  const size_t stride = (size_t)gridDim.x * blockDim.x;
  for (size_t i = (size_t)blockIdx.x * blockDim.x + threadIdx.x; i < total; i += stride) {
    const int r = (int)(i / Kpad), c = (int)(i % Kpad);
    const float v = (r < rows && c < cols) ? S[(size_t)r * cols + c] : 0.f;
    A[i] = (_Float16)v;
  }
}

__global__ __launch_bounds__(256) void k_stats(
    const float* __restrict__ C, int rows, int cout,
    float* __restrict__ sums, float* __restrict__ sumsq) {
  __shared__ float ss[256], sq[256];
  const int ch = blockIdx.x;
  const int tid = threadIdx.x;
  float s = 0.f, q = 0.f;
  for (int r = blockIdx.y * blockDim.x + tid; r < rows;
       r += gridDim.y * blockDim.x) {
    const float x = C[(size_t)r * cout + ch];
    s += x; q += x * x;
  }
  ss[tid] = s; sq[tid] = q;
  __syncthreads();
  for (int st = 128; st > 0; st >>= 1) {
    if (tid < st) { ss[tid] += ss[tid + st]; sq[tid] += sq[tid + st]; }
    __syncthreads();
  }
  if (tid == 0) {
    atomicAdd(&sums[ch], ss[0]);
    atomicAdd(&sumsq[ch], sq[0]);
  }
}

__global__ void k_bn_relu(float* __restrict__ C,
                          const float* __restrict__ sums,
                          const float* __restrict__ sumsq,
                          const float* __restrict__ g,
                          const float* __restrict__ bt,
                          int rows, int cout, float invc) {
  const size_t total = (size_t)rows * cout;
  const size_t stride = (size_t)gridDim.x * blockDim.x;
  for (size_t i = (size_t)blockIdx.x * blockDim.x + threadIdx.x; i < total; i += stride) {
    const int ch = (int)(i % cout);
    const float m = sums[ch] * invc;
    const float v = sumsq[ch] * invc - m * m;
    const float y = g[ch] * (C[i] - m) * rsqrtf(v + EPSV) + bt[ch];
    C[i] = fmaxf(y, 0.f);
  }
}

__global__ void k_maxpool(const float* __restrict__ C, float* __restrict__ P,
                          int groups, int K, int cout) {
  const size_t total = (size_t)groups * cout;
  const size_t stride = (size_t)gridDim.x * blockDim.x;
  for (size_t i = (size_t)blockIdx.x * blockDim.x + threadIdx.x; i < total; i += stride) {
    const int gi = (int)(i / cout), ch = (int)(i % cout);
    float m = -3.4e38f;
    for (int k = 0; k < K; ++k)
      m = fmaxf(m, C[((size_t)gi * K + k) * cout + ch]);
    P[i] = m;
  }
}

// ---------------------------------------------------------------- prototypes + logits
// emb_s/emb_q: (8,5,64). out[i,p,q] = -sum_d (proto[p,d]-emb_q[i,q,d])^2
__global__ __launch_bounds__(320) void k_proto(const float* __restrict__ emb_s,
                                               const float* __restrict__ emb_q,
                                               const int* __restrict__ targets,
                                               float* __restrict__ out) {
  __shared__ float proto[5 * 64];
  const int tid = threadIdx.x;
  if (tid < 320) {
    const int c = tid / 64, d = tid % 64;
    float acc = 0.f;
    for (int i = 0; i < 8; ++i) {
      float s = 0.f; int cnt = 0;
      for (int j = 0; j < 5; ++j) {
        if (targets[i * 5 + j] == c) { ++cnt; s += emb_s[(i * 5 + j) * 64 + d]; }
      }
      acc += s / (cnt > 0 ? (float)cnt : 1.f);
    }
    proto[tid] = acc * 0.125f;
  }
  __syncthreads();
  if (tid < 200) {
    const int i = tid / 25, rem = tid % 25, p = rem / 5, q = rem % 5;
    float s = 0.f;
    for (int d = 0; d < 64; ++d) {
      const float df = proto[p * 64 + d] - emb_q[(i * 5 + q) * 64 + d];
      s += df * df;
    }
    out[tid] = -s;
  }
}

// ================================================================ host
extern "C" void kernel_launch(void* const* d_in, const int* in_sizes, int n_in,
                              void* d_out, int out_size, void* d_ws, size_t ws_size,
                              hipStream_t stream) {
  const float* xyz_in  = (const float*)d_in[0];
  const int*   targets = (const int*)d_in[1];
  const float* xx_in   = (const float*)d_in[2];
  // d_in[3] = targets1 (unused by the reference)
  // params leaves in setup_inputs insertion order:
  //   sa1 (W,b,g,bt)x3 | sa2 x3 | sa3 x3 | fc1 W,b | bn1 g,b | fc2 W,b | bn2 g,b | fc3 W,b
  const float* P[46];
  for (int i = 0; i < 46; ++i) P[i] = (4 + i < n_in) ? (const float*)d_in[4 + i] : nullptr;

  // ---- workspace carve ----
  char* ws = (char*)d_ws;
  size_t off = 0;
  auto carve = [&](size_t bytes) -> char* {
    char* p = ws + off;
    off += (bytes + 255) & ~(size_t)255;
    return p;
  };
  _Float16* Abuf = (_Float16*)carve((size_t)327680 * 160 * 2);   // 105 MB
  float*    Cbuf = (float*)   carve((size_t)655360 * 128 * 4);   // 336 MB
  _Float16* Wh   = (_Float16*)carve((size_t)1024 * 1024 * 2);    // 2 MB
  float*    l1x  = (float*)carve((size_t)40 * 512 * 3 * 4);
  float*    l1p  = (float*)carve((size_t)40 * 512 * 128 * 4);
  float*    l2x  = (float*)carve((size_t)40 * 128 * 3 * 4);
  float*    l2p  = (float*)carve((size_t)40 * 128 * 256 * 4);
  float*    l3p  = (float*)carve((size_t)40 * 1024 * 4);
  float*    stats = (float*)carve(2048 * 4);                      // sums | sumsq
  float*    emb_s = (float*)carve((size_t)40 * 64 * 4);
  float*    emb_q = (float*)carve((size_t)40 * 64 * 4);
  (void)ws_size; (void)in_sizes; (void)out_size;

  auto gemm = [&](const _Float16* Ain, int rows, int rowsPad, int Kpad,
                  const float* W, const float* bias, int cin, int cout,
                  float* Cout) {
    {
      const size_t t = (size_t)cout * Kpad;
      const int nb = (int)((t + 255) / 256);
      k_pack_half<<<nb, 256, 0, stream>>>(W, Wh, cout, cout, cin, Kpad);
    }
    dim3 g((rowsPad / 16 + 15) / 16, cout / 32);
    k_gemm64x32<<<g, 128, 0, stream>>>(Ain, Wh, bias, Cout, rows, rowsPad, Kpad, cout);
  };
  auto bn = [&](float* Cio, int rows, int cout, const float* g, const float* bt) {
    k_zero<<<8, 256, 0, stream>>>(stats, 2048);
    k_stats<<<dim3(cout, 64), 256, 0, stream>>>(Cio, rows, cout, stats, stats + 1024);
    const size_t t = (size_t)rows * cout;
    size_t nb = (t + 255) / 256; if (nb > 8192) nb = 8192;
    k_bn_relu<<<(int)nb, 256, 0, stream>>>(Cio, stats, stats + 1024, g, bt, rows,
                                           cout, 1.f / (float)rows);
  };
  auto pack = [&](const float* Cin, int rows, int rowsPad, int cout) {
    const size_t t = (size_t)rowsPad * cout;
    size_t nb = (t + 255) / 256; if (nb > 16384) nb = 16384;
    k_pack_half<<<(int)nb, 256, 0, stream>>>(Cin, Abuf, rows, rowsPad, cout, cout);
  };
  auto pool = [&](float* dst, int groups, int K, int cout) {
    const size_t t = (size_t)groups * cout;
    size_t nb = (t + 255) / 256; if (nb > 8192) nb = 8192;
    k_maxpool<<<(int)nb, 256, 0, stream>>>(Cbuf, dst, groups, K, cout);
  };

  auto embed = [&](const float* pc, float* emb) {
    const int Bc = 40;
    // ---- SA1: 512 centroids, r=0.2, ns=32, 6 -> 64 -> 64 -> 128
    k_fps<<<Bc, 256, 0, stream>>>(pc, 6, 1024, 512, l1x);
    {
      const int tt = Bc * 512;
      k_ball_group<<<(tt + 255) / 256, 256, 0, stream>>>(
          pc, 6, pc + 3, 6, 3, l1x, Abuf, 0.04f, tt, 1024, 512, 32, 32);
    }
    const int R1 = Bc * 512 * 32;  // 655360
    gemm(Abuf, R1, R1, 32, P[0], P[1], 6, 64, Cbuf);   bn(Cbuf, R1, 64, P[2], P[3]);
    pack(Cbuf, R1, R1, 64);
    gemm(Abuf, R1, R1, 64, P[4], P[5], 64, 64, Cbuf);  bn(Cbuf, R1, 64, P[6], P[7]);
    pack(Cbuf, R1, R1, 64);
    gemm(Abuf, R1, R1, 64, P[8], P[9], 64, 128, Cbuf); bn(Cbuf, R1, 128, P[10], P[11]);
    pool(l1p, Bc * 512, 32, 128);

    // ---- SA2: 128 centroids, r=0.4, ns=64, 131 -> 128 -> 128 -> 256
    k_fps<<<Bc, 256, 0, stream>>>(l1x, 3, 512, 128, l2x);
    {
      const int tt = Bc * 128;
      k_ball_group<<<(tt + 255) / 256, 256, 0, stream>>>(
          l1x, 3, l1p, 128, 128, l2x, Abuf, 0.16f, tt, 512, 128, 64, 160);
    }
    const int R2 = Bc * 128 * 64;  // 327680
    gemm(Abuf, R2, R2, 160, P[12], P[13], 131, 128, Cbuf); bn(Cbuf, R2, 128, P[14], P[15]);
    pack(Cbuf, R2, R2, 128);
    gemm(Abuf, R2, R2, 128, P[16], P[17], 128, 128, Cbuf); bn(Cbuf, R2, 128, P[18], P[19]);
    pack(Cbuf, R2, R2, 128);
    gemm(Abuf, R2, R2, 128, P[20], P[21], 128, 256, Cbuf); bn(Cbuf, R2, 256, P[22], P[23]);
    pool(l2p, Bc * 128, 64, 256);

    // ---- SA3 (group all): 259 -> 256 -> 512 -> 1024, pool over 128 pts
    {
      const int tt = Bc * 128;
      k_groupall<<<(tt + 255) / 256, 256, 0, stream>>>(l2x, l2p, Abuf, tt, 256, 288);
    }
    const int R3 = Bc * 128;  // 5120
    gemm(Abuf, R3, R3, 288, P[24], P[25], 259, 256, Cbuf);  bn(Cbuf, R3, 256, P[26], P[27]);
    pack(Cbuf, R3, R3, 256);
    gemm(Abuf, R3, R3, 256, P[28], P[29], 256, 512, Cbuf);  bn(Cbuf, R3, 512, P[30], P[31]);
    pack(Cbuf, R3, R3, 512);
    gemm(Abuf, R3, R3, 512, P[32], P[33], 512, 1024, Cbuf); bn(Cbuf, R3, 1024, P[34], P[35]);
    pool(l3p, Bc, 128, 1024);

    // ---- FC head: 1024 -> 512 -> 256 -> 64 (rows 40 padded to 48)
    {
      const size_t t = (size_t)48 * 1024;
      k_pack_half<<<(int)((t + 255) / 256), 256, 0, stream>>>(l3p, Abuf, 40, 48, 1024, 1024);
    }
    gemm(Abuf, 40, 48, 1024, P[36], P[37], 1024, 512, Cbuf); bn(Cbuf, 40, 512, P[38], P[39]);
    {
      const size_t t = (size_t)48 * 512;
      k_pack_half<<<(int)((t + 255) / 256), 256, 0, stream>>>(Cbuf, Abuf, 40, 48, 512, 512);
    }
    gemm(Abuf, 40, 48, 512, P[40], P[41], 512, 256, Cbuf);   bn(Cbuf, 40, 256, P[42], P[43]);
    {
      const size_t t = (size_t)48 * 256;
      k_pack_half<<<(int)((t + 255) / 256), 256, 0, stream>>>(Cbuf, Abuf, 40, 48, 256, 256);
    }
    gemm(Abuf, 40, 48, 256, P[44], P[45], 256, 64, emb);     // no BN, bias kept
  };

  embed(xyz_in, emb_s);
  embed(xx_in,  emb_q);
  k_proto<<<1, 320, 0, stream>>>(emb_s, emb_q, targets, (float*)d_out);
}
